// MultiHeadAttnWithRoPE_80745385164978
// MI455X (gfx1250) — matrix operand order repaired and checked by reading, not compiled, not run on hardware
//
#include <hip/hip_runtime.h>
#include <hip/hip_bf16.h>

typedef _Float16 half_t;
typedef __attribute__((ext_vector_type(8)))  _Float16     v8h;
typedef __attribute__((ext_vector_type(16))) _Float16     v16h;
typedef __attribute__((ext_vector_type(8)))  float        v8f;
typedef __attribute__((ext_vector_type(4)))  unsigned int u32x4;
typedef __attribute__((ext_vector_type(8)))  int          i32x8;
typedef __attribute__((ext_vector_type(4)))  int          i32x4;

#define D_MODEL 768
#define SEQ     2048
#define BATCH   4
#define NHEAD   12
#define HDIM    64
#define M_ROWS  (BATCH*SEQ)   // 8192

// Build a v16h A/B fragment from two contiguous 16-byte chunks.
__device__ __forceinline__ v16h mk16(const half_t* p0, const half_t* p1) {
  v8h lo = *(const v8h*)p0;
  v8h hi = *(const v8h*)p1;
  return __builtin_shufflevector(lo, hi, 0,1,2,3,4,5,6,7,8,9,10,11,12,13,14,15);
}

// ---- Tensor Data Mover: async DMA of a 2D tile (rows x 32 f16) global -> LDS.
// D# packing per CDNA5 ISA. Uniform (scalar) operands.
__device__ __forceinline__ void tdm_load_tile(const half_t* gbase, unsigned int ldsAddr,
                                              int rows, int strideElems) {
  unsigned long long ga = (unsigned long long)(uintptr_t)gbase;
  u32x4 g0;
  g0[0] = 1u;                                   // count=1, is_restore=0, gather off
  g0[1] = ldsAddr;                              // lds_addr (bytes)
  g0[2] = (unsigned int)ga;                     // global_addr[31:0]
  g0[3] = (unsigned int)((ga >> 32) & 0x01FFFFFFu) | (2u << 30);  // addr[56:32] | type=2
  i32x8 g1;
  g1[0] = 0x00010000;                           // workgroup_mask=0, data_size=1 (2 bytes)
  g1[1] = 0;                                    // tensor_dim0 = 1<<20 -> low16 = 0
  g1[2] = 0x10;                                 // tensor_dim0 hi16=0x10 ; tensor_dim1 low16=0
  g1[3] = 0x10 | (32 << 16);                    // tensor_dim1 hi16=0x10 ; tile_dim0=32
  g1[4] = rows & 0xFFFF;                        // tile_dim1=rows ; tile_dim2=0
  g1[5] = strideElems;                          // tensor_dim0_stride[31:0]
  g1[6] = 0;                                    // stride0 hi ; stride1 low16
  g1[7] = 0;                                    // stride1 hi
  i32x4 z4 = {0, 0, 0, 0};
#if defined(__clang_major__) && (__clang_major__ >= 23)
  i32x8 z8 = {0, 0, 0, 0, 0, 0, 0, 0};
  __builtin_amdgcn_tensor_load_to_lds(g0, g1, z4, z4, z8, 0);
#else
  __builtin_amdgcn_tensor_load_to_lds(g0, g1, z4, z4, 0);
#endif
}

__global__ void cvt_f32_f16(const float* __restrict__ in, half_t* __restrict__ out, int n) {
  int i = blockIdx.x * blockDim.x + threadIdx.x;
  if (i < n) out[i] = (half_t)in[i];
}

// Wt[n*K + k] = (f16) W[k*N + n]   (coalesced writes)
__global__ void wtrans_f16(const float* __restrict__ W, half_t* __restrict__ Wt, int K, int N) {
  int i = blockIdx.x * blockDim.x + threadIdx.x;
  if (i < K * N) {
    int n = i / K;
    int k = i % K;
    Wt[i] = (half_t)W[(size_t)k * N + n];
  }
}

// C[MxN] = A[MxK](f16,row-major) @ Bt[NxK](f16,row-major)^T + bias
// Tiles staged to LDS by the Tensor Data Mover, double-buffered.
// EPI 0: f32 out row-major.  EPI 1: RoPE then f16 [B,H,S,hd].  EPI 2: f16 [B,H,hd,S].
template<int EPI>
__global__ __launch_bounds__(256)
void gemm_wmma(const half_t* __restrict__ A, const half_t* __restrict__ Bt,
               const float* __restrict__ bias, half_t* __restrict__ outH,
               float* __restrict__ outF, int M, int N, int K) {
  __shared__ half_t tA[2][128 * 32];   // 8 KB x2
  __shared__ half_t tB[2][64 * 32];    // 4 KB x2

  const int lane = threadIdx.x & 31;
  const int wave = threadIdx.x >> 5;
  const int l16  = lane & 15;
  const int lh   = lane >> 4;
  const int rowTile = blockIdx.y * 128;
  const int nBase   = blockIdx.x * 64;

  v8f acc[4];
  #pragma unroll
  for (int f = 0; f < 4; ++f) {
    float bvv = bias[nBase + f * 16 + l16];
    #pragma unroll
    for (int r = 0; r < 8; ++r) acc[f][r] = bvv;
  }

  const int kSteps = K >> 5;
  // Prologue: stage k-chunk 0 into buffer 0.
  if (wave == 0)
    tdm_load_tile(A + (size_t)rowTile * K, (unsigned int)(uintptr_t)&tA[0][0], 128, K);
  if (wave == 1)
    tdm_load_tile(Bt + (size_t)nBase * K, (unsigned int)(uintptr_t)&tB[0][0], 64, K);

  for (int kc = 0; kc < kSteps; ++kc) {
    const int buf = kc & 1;
    if (kc + 1 < kSteps) {
      const int kb2 = (kc + 1) * 32;
      if (wave == 0)
        tdm_load_tile(A + (size_t)rowTile * K + kb2,
                      (unsigned int)(uintptr_t)&tA[buf ^ 1][0], 128, K);
      if (wave == 1)
        tdm_load_tile(Bt + (size_t)nBase * K + kb2,
                      (unsigned int)(uintptr_t)&tB[buf ^ 1][0], 64, K);
      if (wave < 2) __builtin_amdgcn_s_wait_tensorcnt(1);  // chunk kc landed
    } else {
      if (wave < 2) __builtin_amdgcn_s_wait_tensorcnt(0);
    }
    __syncthreads();

    const half_t* ar = &tA[buf][(wave * 16 + l16) * 32];
    v16h af = mk16(ar + lh * 16, ar + lh * 16 + 8);
    #pragma unroll
    for (int f = 0; f < 4; ++f) {
      const half_t* br = &tB[buf][(f * 16 + l16) * 32 + lh * 16];
      v16h bf = mk16(br, br + 8);
      acc[f] = __builtin_amdgcn_wmma_f32_16x16x32_f16(
          false, af, false, bf, (short)0, acc[f], false, false);
    }
    __syncthreads();
  }

  #pragma unroll
  for (int f = 0; f < 4; ++f) {
    const int n = nBase + f * 16 + l16;
    #pragma unroll
    for (int r = 0; r < 8; ++r) {
      const int m = rowTile + wave * 16 + r + lh * 8;  // C layout: VGPR r -> row r + lh*8
      float v = acc[f][r];
      if (EPI == 0) {
        outF[(size_t)m * N + n] = v;
      } else if (EPI == 1) {
        // RoPE over full D: pair (2p, 2p+1) sits in adjacent lanes.
        float vp = __shfl_xor(v, 1, 32);
        int   p  = n >> 1;
        float inv_f = __expf(-(2.0f * (float)p / (float)D_MODEL) * 9.210340371976184f); // ln(1e4)
        int b = m >> 11, s = m & (SEQ - 1);
        float sn, cs;
        __sincosf((float)s * inv_f, &sn, &cs);
        float ov = (n & 1) ? (vp * sn + v * cs) : (v * cs - vp * sn);
        int h = n >> 6, e = n & 63;
        outH[(((size_t)b * NHEAD + h) * SEQ + s) * HDIM + e] = (half_t)ov;
      } else {
        int b = m >> 11, s = m & (SEQ - 1);
        int h = n >> 6, e = n & 63;
        outH[(((size_t)b * NHEAD + h) * HDIM + e) * SEQ + s] = (half_t)v;
      }
    }
  }
}

// One wave per (b, h, 16-query tile). Streaming causal flash attention,
// 32 keys per chunk so the P@V WMMA uses its full K dimension.
__global__ __launch_bounds__(32)
void flash_attn(const half_t* __restrict__ Qh, const half_t* __restrict__ Kh,
                const half_t* __restrict__ Vt, half_t* __restrict__ Ah) {
  const int lane = threadIdx.x & 31;
  const int l16  = lane & 15;
  const int lh   = lane >> 4;
  const int idx  = blockIdx.x;
  const int qt = idx & 127;                 // S/16 tiles
  const int h  = (idx >> 7) % NHEAD;
  const int b  = idx / (128 * NHEAD);
  const int bh = b * NHEAD + h;

  __shared__ half_t sP[16 * 40];            // 16x32 P tile, row stride 40 halves (16B-aligned)

  const half_t* qrow = Qh + ((size_t)bh * SEQ + qt * 16 + l16) * HDIM;
  v16h qa0 = mk16(qrow + lh * 16,      qrow + lh * 16 + 8);
  v16h qa1 = mk16(qrow + 32 + lh * 16, qrow + 32 + lh * 16 + 8);

  v8f o[4] = {};
  float mrow[8], lrow[8];
  #pragma unroll
  for (int r = 0; r < 8; ++r) { mrow[r] = -__builtin_inff(); lrow[r] = 0.0f; }

  const int nKeys   = (qt + 1) * 16;
  const int nChunks = (nKeys + 31) >> 5;

  for (int c = 0; c < nChunks; ++c) {
    const int k0 = c * 32;
    const half_t* krowA = Kh + ((size_t)bh * SEQ + k0 + l16) * HDIM;
    const half_t* krowB = krowA + (size_t)16 * HDIM;
    v16h kbA0 = mk16(krowA + lh * 16,      krowA + lh * 16 + 8);
    v16h kbA1 = mk16(krowA + 32 + lh * 16, krowA + 32 + lh * 16 + 8);
    v16h kbB0 = mk16(krowB + lh * 16,      krowB + lh * 16 + 8);
    v16h kbB1 = mk16(krowB + 32 + lh * 16, krowB + 32 + lh * 16 + 8);

    v8f s0 = {}, s1 = {};
    s0 = __builtin_amdgcn_wmma_f32_16x16x32_f16(false, qa0, false, kbA0, (short)0, s0, false, false);
    s0 = __builtin_amdgcn_wmma_f32_16x16x32_f16(false, qa1, false, kbA1, (short)0, s0, false, false);
    s1 = __builtin_amdgcn_wmma_f32_16x16x32_f16(false, qa0, false, kbB0, (short)0, s1, false, false);
    s1 = __builtin_amdgcn_wmma_f32_16x16x32_f16(false, qa1, false, kbB1, (short)0, s1, false, false);

    const bool last = (c == nChunks - 1);   // only the last chunk can cross the diagonal
    float p0[8], p1[8];
    #pragma unroll
    for (int r = 0; r < 8; ++r) {
      float sa = s0[r] * 0.125f;            // 1/sqrt(64)
      float sb = s1[r] * 0.125f;
      if (last) {
        int qrowg = qt * 16 + r + lh * 8;   // global query row; key cols: k0+l16, k0+16+l16
        if (k0 + l16      > qrowg) sa = -__builtin_inff();
        if (k0 + 16 + l16 > qrowg) sb = -__builtin_inff();
      }
      float rm = fmaxf(sa, sb);
      #pragma unroll
      for (int off = 1; off < 16; off <<= 1) rm = fmaxf(rm, __shfl_xor(rm, off, 32));
      float nm   = fmaxf(mrow[r], rm);
      float corr = __expf(mrow[r] - nm);
      float pa_  = __expf(sa - nm);
      float pb_  = __expf(sb - nm);
      float rs   = pa_ + pb_;
      #pragma unroll
      for (int off = 1; off < 16; off <<= 1) rs += __shfl_xor(rs, off, 32);
      lrow[r] = lrow[r] * corr + rs;
      mrow[r] = nm;
      p0[r] = pa_;
      p1[r] = pb_;
      #pragma unroll
      for (int f = 0; f < 4; ++f) o[f][r] *= corr;
    }

    // C-layout -> A-layout transpose of the 16x32 P tile through LDS.
    __syncthreads();
    #pragma unroll
    for (int r = 0; r < 8; ++r) {
      sP[(r + lh * 8) * 40 + l16]      = (half_t)p0[r];
      sP[(r + lh * 8) * 40 + 16 + l16] = (half_t)p1[r];
    }
    __syncthreads();
    v16h pa = mk16(&sP[l16 * 40 + lh * 16], &sP[l16 * 40 + lh * 16 + 8]);

    const half_t* vbase = Vt + (size_t)bh * HDIM * SEQ + (size_t)k0;
    #pragma unroll
    for (int f = 0; f < 4; ++f) {
      const half_t* vrow = vbase + (size_t)(f * 16 + l16) * SEQ + lh * 16;
      v16h vb = mk16(vrow, vrow + 8);
      o[f] = __builtin_amdgcn_wmma_f32_16x16x32_f16(false, pa, false, vb, (short)0, o[f], false, false);
    }
  }

  #pragma unroll
  for (int r = 0; r < 8; ++r) {
    float inv = 1.0f / lrow[r];
    int s = qt * 16 + r + lh * 8;
    #pragma unroll
    for (int f = 0; f < 4; ++f) {
      int e = f * 16 + l16;
      Ah[((size_t)b * SEQ + s) * D_MODEL + h * HDIM + e] = (half_t)(o[f][r] * inv);
    }
  }
}

extern "C" void kernel_launch(void* const* d_in, const int* in_sizes, int n_in,
                              void* d_out, int out_size, void* d_ws, size_t ws_size,
                              hipStream_t stream) {
  const float* X  = (const float*)d_in[0];
  const float* Wq = (const float*)d_in[1];
  const float* bq = (const float*)d_in[2];
  const float* Wk = (const float*)d_in[3];
  const float* bk = (const float*)d_in[4];
  const float* Wv = (const float*)d_in[5];
  const float* bv = (const float*)d_in[6];
  const float* Wo = (const float*)d_in[7];
  const float* bo = (const float*)d_in[8];
  float* out = (float*)d_out;

  const size_t MK = (size_t)M_ROWS * D_MODEL;   // 6,291,456
  const size_t KN = (size_t)D_MODEL * D_MODEL;  //   589,824
  half_t* ws  = (half_t*)d_ws;
  half_t* Xh  = ws;
  half_t* Wqt = Xh  + MK;
  half_t* Wkt = Wqt + KN;
  half_t* Wvt = Wkt + KN;
  half_t* Wot = Wvt + KN;
  half_t* Qh  = Wot + KN;
  half_t* Kh  = Qh  + MK;
  half_t* Vth = Kh  + MK;
  half_t* Ah  = Vth + MK;

  cvt_f32_f16<<<(int)((MK + 255) / 256), 256, 0, stream>>>(X, Xh, (int)MK);
  const int wblk = (int)((KN + 255) / 256);
  wtrans_f16<<<wblk, 256, 0, stream>>>(Wq, Wqt, D_MODEL, D_MODEL);
  wtrans_f16<<<wblk, 256, 0, stream>>>(Wk, Wkt, D_MODEL, D_MODEL);
  wtrans_f16<<<wblk, 256, 0, stream>>>(Wv, Wvt, D_MODEL, D_MODEL);
  wtrans_f16<<<wblk, 256, 0, stream>>>(Wo, Wot, D_MODEL, D_MODEL);

  dim3 ggrid(D_MODEL / 64, M_ROWS / 128);
  gemm_wmma<1><<<ggrid, 256, 0, stream>>>(Xh, Wqt, bq, Qh,  nullptr, M_ROWS, D_MODEL, D_MODEL);
  gemm_wmma<1><<<ggrid, 256, 0, stream>>>(Xh, Wkt, bk, Kh,  nullptr, M_ROWS, D_MODEL, D_MODEL);
  gemm_wmma<2><<<ggrid, 256, 0, stream>>>(Xh, Wvt, bv, Vth, nullptr, M_ROWS, D_MODEL, D_MODEL);

  flash_attn<<<BATCH * NHEAD * (SEQ / 16), 32, 0, stream>>>(Qh, Kh, Vth, Ah);

  gemm_wmma<0><<<ggrid, 256, 0, stream>>>(Ah, Wot, bo, nullptr, out, M_ROWS, D_MODEL, D_MODEL);
}
